// RESNET20MULT3_18674517803242
// MI455X (gfx1250) — matrix operand order repaired and checked by reading, not compiled, not run on hardware
//
#include <hip/hip_runtime.h>
#include <hip/hip_bf16.h>
#include <stdint.h>

// ---------------------------------------------------------------------------
// Types for WMMA fragments / TDM descriptors (gfx1250, wave32)
// ---------------------------------------------------------------------------
typedef __attribute__((ext_vector_type(16))) __bf16 v16bf;
typedef __attribute__((ext_vector_type(8)))  float  v8f;
typedef __attribute__((ext_vector_type(4)))  unsigned int v4u;
typedef __attribute__((ext_vector_type(8)))  unsigned int v8u;

union FragBF {
  v4u   q[2];   // two 16-byte halves
  v16bf bf;     // 16 bf16 elements
};

__device__ __forceinline__ unsigned short f2bf(float f) {
  unsigned u = __builtin_bit_cast(unsigned, f);
  u += 0x7FFFu + ((u >> 16) & 1u);           // round-to-nearest-even
  return (unsigned short)(u >> 16);
}
__device__ __forceinline__ float bf2f(unsigned short s) {
  return __builtin_bit_cast(float, (unsigned)s << 16);
}

#define EPSV 1e-5f

// ---------------------------------------------------------------------------
// Implicit-GEMM 3x3 SAME conv, stride 1.
//   act : bf16 NHWC, channels padded to CPAD (zeros in pad lanes)
//   wpk : bf16 [Cout][9 taps][CPAD]   (K contiguous per output channel)
//   out : fp32 NHWC, C = Cout (= NT*16)
//
// Weights are DMA'd global->LDS once per workgroup with the Tensor Data Mover
// (tensor_load_to_lds + s_wait_tensorcnt), then B fragments come from LDS.
// One wave computes a 16-pixel x Cout tile via v_wmma_f32_16x16x32_bf16.
//
// A layout (16-bit A 16x32): lane<16 -> K 0..7 / 16..23 ; lane>=16 -> K 8..15 / 24..31
// B layout (16-bit B 32x16): col = lane&15 ; lane<16 -> K 0..7 / 16..23 ; lane>=16 -> K 8..15 / 24..31
// C layout: col = lane&15 ; VGPR j -> row j (+8 for lanes 16..31)
// ---------------------------------------------------------------------------
template<int CPAD, int NT>
__global__ __launch_bounds__(256) void conv3x3_wmma(
    const unsigned short* __restrict__ act,
    const unsigned short* __restrict__ wpk,
    float* __restrict__ out, int H, int W, int Cout)
{
  const int KTOT  = 9 * CPAD;
  const int NELEM = NT * 16 * 9 * CPAD;        // packed-weight elements (bf16)
  __shared__ __align__(16) unsigned short wlds[NT * 16 * 9 * CPAD];

  // --- TDM: DMA packed weights (global -> LDS). Wave 0 issues; all waves sync.
  if ((threadIdx.x >> 5) == 0) {
    unsigned long long ga = (unsigned long long)wpk;
    v4u g0;
    g0[0] = 1u;                                      // count = 1 (valid D#)
    g0[1] = 0u;                                      // lds_addr: wlds is the only LDS object -> offset 0
    g0[2] = (unsigned)ga;                            // global_addr[31:0]
    g0[3] = ((unsigned)(ga >> 32) & 0x01FFFFFFu)     // global_addr[56:32]
            | (2u << 30);                            // type = 2 ("image")
    v8u g1;
    g1[0] = 0x00010000u;                             // wg_mask=0, data_size=1 (2 bytes)
    g1[1] = ((unsigned)NELEM & 0xFFFFu) << 16;       // tensor_dim0[15:0]  (bits 63:48)
    g1[2] = ((unsigned)NELEM >> 16) | (1u << 16);    // tensor_dim0[31:16]; tensor_dim1 = 1
    g1[3] = ((unsigned)NELEM & 0xFFFFu) << 16;       // tile_dim0 = NELEM (bits 127:112)
    g1[4] = 0u;                                      // tile_dim1 / tile_dim2 unused
    g1[5] = (unsigned)NELEM;                         // tensor_dim0_stride[31:0]
    g1[6] = 0u;                                      // stride hi / dim1_stride lo
    g1[7] = 0u;                                      // dim1_stride hi
    asm volatile("tensor_load_to_lds %0, %1" :: "s"(g0), "s"(g1) : "memory");
    __builtin_amdgcn_s_wait_tensorcnt(0);
  }
  __syncthreads();

  const int lane = threadIdx.x & 31;
  const int wave = (blockIdx.x << 3) + (threadIdx.x >> 5);
  const int r    = lane & 15;
  const int hi   = lane >> 4;
  const int HW   = H * W;

  long p   = (long)wave * 16 + r;      // this lane's pixel (A row)
  int  n   = (int)(p / HW);
  int  rem = (int)(p % HW);
  int  y   = rem / W;
  int  x   = rem % W;

  v8f acc[NT];
#pragma unroll
  for (int t = 0; t < NT; ++t)
#pragma unroll
    for (int j = 0; j < 8; ++j) acc[t][j] = 0.0f;

  const v4u vz = {0u, 0u, 0u, 0u};

#pragma unroll
  for (int tap = 0; tap < 9; ++tap) {
    const int iy = y + tap / 3 - 1;
    const int ix = x + tap % 3 - 1;
    const bool valid = (iy >= 0) & (iy < H) & (ix >= 0) & (ix < W);
    const unsigned short* abase = act + (((long)n * H + iy) * W + ix) * CPAD;
    // Speculative SYS-scope prefetch: silently dropped if halo address is bad.
    __builtin_prefetch(abase + CPAD, 0, 0);

#pragma unroll
    for (int ks = 0; ks < CPAD / 32; ++ks) {
      FragBF fa;
      const int ko = ks * 32 + hi * 8;
      if (valid) {
        fa.q[0] = *(const v4u*)(abase + ko);
        fa.q[1] = *(const v4u*)(abase + ko + 16);
      } else {
        fa.q[0] = vz;
        fa.q[1] = vz;
      }
#pragma unroll
      for (int t = 0; t < NT; ++t) {
        FragBF fb;
        const unsigned short* bbase =
            wlds + (t * 16 + r) * KTOT + tap * CPAD + ks * 32 + hi * 8;
        fb.q[0] = *(const v4u*)(bbase);          // ds_load_b128
        fb.q[1] = *(const v4u*)(bbase + 16);     // ds_load_b128
        acc[t] = __builtin_amdgcn_wmma_f32_16x16x32_bf16(
            false, fa.bf, false, fb.bf, (short)0, acc[t], false, false);
      }
    }
  }

  const long rowbase = (long)wave * 16 + hi * 8;
#pragma unroll
  for (int t = 0; t < NT; ++t) {
    const int o = t * 16 + r;
#pragma unroll
    for (int j = 0; j < 8; ++j)
      out[(rowbase + j) * Cout + o] = acc[t][j];
  }
}

// ---------------------------------------------------------------------------
// Pack input NCHW fp32 (1024,3,32,32) -> bf16 NHWC CPAD=32 (zero padded)
// ---------------------------------------------------------------------------
__global__ void pack_x_k(const float* __restrict__ x, unsigned short* __restrict__ act)
{
  long i = (long)blockIdx.x * 256 + threadIdx.x;     // over 1024*1024*32
  const long total = 1024L * 1024 * 32;
  if (i >= total) return;
  int  c = (int)(i & 31);
  long p = i >> 5;                  // n*1024 + y*32 + x
  float v = 0.0f;
  if (c < 3) {
    long n   = p >> 10;
    long rem = p & 1023;
    v = x[(n * 3 + c) * 1024 + rem];
  }
  act[i] = f2bf(v);
}

// ---------------------------------------------------------------------------
// Pack weights OIHW fp32 -> bf16 [Cout][9][CPAD] (zero pad Cin -> CPAD)
// ---------------------------------------------------------------------------
__global__ void pack_w_k(const float* __restrict__ w, unsigned short* __restrict__ wpk,
                         int Cout, int Cin, int CP)
{
  int total = Cout * 9 * CP;
  int i = blockIdx.x * 256 + threadIdx.x;
  if (i >= total) return;
  int c = i % CP;
  int t = (i / CP) % 9;
  int o = i / (CP * 9);
  float v = (c < Cin) ? w[(o * Cin + c) * 9 + t] : 0.0f;
  wpk[i] = f2bf(v);
}

__global__ void zero_stats_k(float* s) { if (threadIdx.x < 128) s[threadIdx.x] = 0.0f; }

// ---------------------------------------------------------------------------
// Per-channel sum / sumsq over fp32 NHWC tensor. grid = C * NSLICE blocks.
// ---------------------------------------------------------------------------
__global__ __launch_bounds__(256) void stats_k(const float* __restrict__ in,
                                               float* __restrict__ stats,
                                               long npix, int C)
{
  const int c     = blockIdx.x % C;
  const int slice = blockIdx.x / C;
  const int NS    = gridDim.x / C;
  float s = 0.0f, s2 = 0.0f;
  for (long p = (long)slice * 256 + threadIdx.x; p < npix; p += (long)NS * 256) {
    float v = in[p * C + c];
    s += v; s2 += v * v;
  }
  __shared__ float sh0[256], sh1[256];
  sh0[threadIdx.x] = s; sh1[threadIdx.x] = s2;
  __syncthreads();
  for (int st = 128; st > 0; st >>= 1) {
    if ((int)threadIdx.x < st) {
      sh0[threadIdx.x] += sh0[threadIdx.x + st];
      sh1[threadIdx.x] += sh1[threadIdx.x + st];
    }
    __syncthreads();
  }
  if (threadIdx.x == 0) {
    atomicAdd(&stats[2 * c],     sh0[0]);
    atomicAdd(&stats[2 * c + 1], sh1[0]);
  }
}

// ---------------------------------------------------------------------------
// 2x2 maxpool stride 2, fp32 NHWC
// ---------------------------------------------------------------------------
__global__ void maxpool2_k(const float* __restrict__ in, float* __restrict__ out,
                           int N, int H, int W, int C)
{
  const int Ho = H >> 1, Wo = W >> 1;
  long total = (long)N * Ho * Wo * C;
  long i = (long)blockIdx.x * 256 + threadIdx.x;
  if (i >= total) return;
  int  c  = (int)(i % C);
  long p  = i / C;
  int  xo = (int)(p % Wo);
  long t  = p / Wo;
  int  yo = (int)(t % Ho);
  int  n  = (int)(t / Ho);
  const float* b = in + (((long)n * H + 2 * yo) * W + 2 * xo) * C + c;
  float v = fmaxf(fmaxf(b[0], b[C]), fmaxf(b[(long)W * C], b[(long)W * C + C]));
  out[i] = v;
}

// ---------------------------------------------------------------------------
// BN(batch stats) + ReLU + repack to bf16 NHWC with channel padding
// ---------------------------------------------------------------------------
__global__ void bn_relu_pack_k(const float* __restrict__ in, const float* __restrict__ stats,
                               const float* __restrict__ g, const float* __restrict__ b,
                               unsigned short* __restrict__ outact,
                               long npix, int C, int CP, float invcnt)
{
  long total = npix * CP;
  long i = (long)blockIdx.x * 256 + threadIdx.x;
  if (i >= total) return;
  int  c = (int)(i % CP);
  long p = i / CP;
  float v = 0.0f;
  if (c < C) {
    float m   = stats[2 * c] * invcnt;
    float var = stats[2 * c + 1] * invcnt - m * m;
    float xin = in[p * C + c];
    v = (xin - m) * rsqrtf(var + EPSV) * g[c] + b[c];
    v = v > 0.0f ? v : 0.0f;
  }
  outact[i] = f2bf(v);
}

// ---------------------------------------------------------------------------
// BN + residual add (+ downsample shortcut: stride-2 slice, channel pad) + ReLU
// Writes next-stage bf16 NHWC padded activations.
// ---------------------------------------------------------------------------
__global__ void residual_pack_k(const float* __restrict__ t2, const float* __restrict__ stats,
                                const float* __restrict__ g, const float* __restrict__ b,
                                const unsigned short* __restrict__ actin,
                                unsigned short* __restrict__ outact,
                                int N, int Ho, int Wo, int C, int CP,
                                int CinReal, int CPin, int down, float invcnt)
{
  long total = (long)N * Ho * Wo * CP;
  long i = (long)blockIdx.x * 256 + threadIdx.x;
  if (i >= total) return;
  int  c  = (int)(i % CP);
  long p  = i / CP;
  int  xo = (int)(p % Wo);
  long t  = p / Wo;
  int  yo = (int)(t % Ho);
  int  n  = (int)(t / Ho);
  float v = 0.0f;
  if (c < C) {
    float m   = stats[2 * c] * invcnt;
    float var = stats[2 * c + 1] * invcnt - m * m;
    float o   = (t2[p * C + c] - m) * rsqrtf(var + EPSV) * g[c] + b[c];
    float sc  = 0.0f;
    if (!down) {
      sc = bf2f(actin[p * CPin + c]);
    } else {
      int d  = C - CinReal;
      int cs = c - d / 2;
      if (cs >= 0 && cs < CinReal)
        sc = bf2f(actin[(((long)n * (2 * Ho) + 2 * yo) * (2 * Wo) + 2 * xo) * CPin + cs]);
    }
    v = o + sc;
    v = v > 0.0f ? v : 0.0f;
  }
  outact[i] = f2bf(v);
}

// ---------------------------------------------------------------------------
// Global average pool (8x8) + FC 64 -> 10. One block per image.
// ---------------------------------------------------------------------------
__global__ __launch_bounds__(256) void avgfc_k(const unsigned short* __restrict__ act,
                                               const float* __restrict__ fcw,
                                               const float* __restrict__ fcb,
                                               float* __restrict__ out)
{
  const int n = blockIdx.x;
  __shared__ float h[64];
  const int t = threadIdx.x;
  if (t < 64) {
    float s = 0.0f;
    for (int p = 0; p < 64; ++p)
      s += bf2f(act[((long)n * 64 + p) * 64 + t]);   // H=W=8, CP=64
    h[t] = s * (1.0f / 64.0f);
  }
  __syncthreads();
  if (t < 10) {
    float s = fcb[t];
    for (int c = 0; c < 64; ++c) s += h[c] * fcw[t * 64 + c];
    out[n * 10 + t] = s;
  }
}

// ---------------------------------------------------------------------------
// Host orchestration
// ---------------------------------------------------------------------------
static inline int padc(int c) { return c <= 32 ? 32 : 64; }

static void launch_conv(int CPAD, int NT, const unsigned short* act, const unsigned short* wpk,
                        float* out, int H, int W, int Cout, hipStream_t s)
{
  long NHW = 1024L * H * W;
  int blocks = (int)(NHW / 128);       // 8 waves/block * 16 pixels/wave
  if      (CPAD == 32 && NT == 1) conv3x3_wmma<32, 1><<<blocks, 256, 0, s>>>(act, wpk, out, H, W, Cout);
  else if (CPAD == 32 && NT == 2) conv3x3_wmma<32, 2><<<blocks, 256, 0, s>>>(act, wpk, out, H, W, Cout);
  else if (CPAD == 32 && NT == 4) conv3x3_wmma<32, 4><<<blocks, 256, 0, s>>>(act, wpk, out, H, W, Cout);
  else                            conv3x3_wmma<64, 4><<<blocks, 256, 0, s>>>(act, wpk, out, H, W, Cout);
}

struct BCfg { int cin, cout, stride, Hin; };

extern "C" void kernel_launch(void* const* d_in, const int* in_sizes, int n_in,
                              void* d_out, int out_size, void* d_ws, size_t ws_size,
                              hipStream_t stream)
{
  (void)in_sizes; (void)n_in; (void)out_size; (void)ws_size;

  // --- input unpack (setup_inputs dict order, leaves flattened) ---
  const float* x      = (const float*)d_in[0];
  const float* stem_w = (const float*)d_in[1];
  const float* stem_g = (const float*)d_in[2];
  const float* stem_b = (const float*)d_in[3];
  const float* fc_w   = (const float*)d_in[58];
  const float* fc_b   = (const float*)d_in[59];

  // --- workspace carve-up ---
  char* ws = (char*)d_ws;
  unsigned short* actA = (unsigned short*)(ws);                      //  67,108,864 B
  unsigned short* actB = (unsigned short*)(ws + 67108864);           //  67,108,864 B
  float*          t1   = (float*)(ws + 134217728);                   // 134,217,728 B
  float*          t1p  = (float*)(ws + 268435456);                   //  33,554,432 B
  unsigned short* wpk  = (unsigned short*)(ws + 301989888);          //     131,072 B
  float*          stats= (float*)(ws + 302120960);                   //         512 B

  static const BCfg cfg[9] = {
    {16, 16, 1, 32}, {16, 16, 1, 32}, {16, 16, 1, 32},
    {16, 32, 2, 32}, {32, 32, 1, 16}, {32, 32, 1, 16},
    {32, 64, 2, 16}, {64, 64, 1,  8}, {64, 64, 1,  8}};

  // --- pack input to bf16 NHWC pad32 ---
  {
    long total = 1024L * 1024 * 32;
    pack_x_k<<<(int)((total + 255) / 256), 256, 0, stream>>>(x, actA);
  }

  // --- stem: conv 3->16, BN, ReLU ---
  {
    int Cout = 16, CPin = 32, CPout = 32;
    long npix = 1024L * 32 * 32;
    pack_w_k<<<(Cout * 9 * CPin + 255) / 256, 256, 0, stream>>>(stem_w, wpk, Cout, 3, CPin);
    launch_conv(CPin, Cout / 16, actA, wpk, t1, 32, 32, Cout, stream);
    zero_stats_k<<<1, 128, 0, stream>>>(stats);
    stats_k<<<Cout * 64, 256, 0, stream>>>(t1, stats, npix, Cout);
    long total = npix * CPout;
    bn_relu_pack_k<<<(int)((total + 255) / 256), 256, 0, stream>>>(
        t1, stats, stem_g, stem_b, actB, npix, Cout, CPout, 1.0f / (float)npix);
  }

  unsigned short* cur = actB;   // current stage activations (bf16 NHWC padded)
  unsigned short* nxt = actA;

  for (int i = 0; i < 9; ++i) {
    const BCfg& B = cfg[i];
    const float* w1 = (const float*)d_in[4 + 6 * i + 0];
    const float* w2 = (const float*)d_in[4 + 6 * i + 1];
    const float* g1 = (const float*)d_in[4 + 6 * i + 2];
    const float* b1 = (const float*)d_in[4 + 6 * i + 3];
    const float* g2 = (const float*)d_in[4 + 6 * i + 4];
    const float* b2 = (const float*)d_in[4 + 6 * i + 5];

    const int Hin  = B.Hin, Hout = B.Hin / B.stride;
    const int CPin = padc(B.cin), CPout = padc(B.cout);
    const long npix_out = 1024L * Hout * Hout;
    const float invcnt  = 1.0f / (float)npix_out;

    // conv1 (at input resolution), then optional pool, stats, BN+ReLU -> mid
    pack_w_k<<<(B.cout * 9 * CPin + 255) / 256, 256, 0, stream>>>(w1, wpk, B.cout, B.cin, CPin);
    launch_conv(CPin, B.cout / 16, cur, wpk, t1, Hin, Hin, B.cout, stream);
    float* convout = t1;
    if (B.stride == 2) {
      long tot = npix_out * B.cout;
      maxpool2_k<<<(int)((tot + 255) / 256), 256, 0, stream>>>(t1, t1p, 1024, Hin, Hin, B.cout);
      convout = t1p;
    }
    zero_stats_k<<<1, 128, 0, stream>>>(stats);
    stats_k<<<B.cout * 64, 256, 0, stream>>>(convout, stats, npix_out, B.cout);
    {
      long tot = npix_out * CPout;
      bn_relu_pack_k<<<(int)((tot + 255) / 256), 256, 0, stream>>>(
          convout, stats, g1, b1, nxt, npix_out, B.cout, CPout, invcnt);
    }

    // conv2 (at output resolution), stats, BN + residual + ReLU -> new stage act
    pack_w_k<<<(B.cout * 9 * CPout + 255) / 256, 256, 0, stream>>>(w2, wpk, B.cout, B.cout, CPout);
    launch_conv(CPout, B.cout / 16, nxt, wpk, t1, Hout, Hout, B.cout, stream);
    zero_stats_k<<<1, 128, 0, stream>>>(stats);
    stats_k<<<B.cout * 64, 256, 0, stream>>>(t1, stats, npix_out, B.cout);

    int down = (B.stride != 1 || B.cin != B.cout) ? 1 : 0;
    {
      long tot = npix_out * CPout;
      residual_pack_k<<<(int)((tot + 255) / 256), 256, 0, stream>>>(
          t1, stats, g2, b2, cur, nxt, 1024, Hout, Hout, B.cout, CPout,
          B.cin, CPin, down, invcnt);
    }
    // nxt now holds the block output (mid buffer was dead after conv2); swap
    unsigned short* tmp = cur; cur = nxt; nxt = tmp;
  }

  // --- head: global avg pool + FC ---
  avgfc_k<<<1024, 256, 0, stream>>>(cur, fc_w, fc_b, (float*)d_out);
}
